// ContextualGatingCollapse_18356690223117
// MI455X (gfx1250) — matrix operand order repaired and checked by reading, not compiled, not run on hardware
//
#include <hip/hip_runtime.h>
#include <hip/hip_bf16.h>
#include <math.h>

// ---------------------------------------------------------------------------
// Problem constants (from reference): B=16, S=4096, D=1024, H=16, HD=64
// ---------------------------------------------------------------------------
#define BATCH 16
#define SEQ   4096
#define DIM   1024
#define HEADS 16
#define HD    64
#define EPSF  1e-6f

typedef __bf16 bf16_t;
typedef __attribute__((ext_vector_type(16))) __bf16 v16bf;
typedef __attribute__((ext_vector_type(8)))  __bf16 v8bf;
typedef __attribute__((ext_vector_type(8)))  float  v8f;

static __device__ __forceinline__ v16bf cat8(v8bf lo, v8bf hi) {
  return __builtin_shufflevector(lo, hi, 0,1,2,3,4,5,6,7,8,9,10,11,12,13,14,15);
}

// elu(x)+1 : x>0 ? x+1 : exp(x)
static __device__ __forceinline__ float phi_map(float x) {
  return x > 0.f ? x + 1.f : __expf(x);
}

// Async global->LDS 16-byte copy (CDNA5; tracked by ASYNCcnt).
// dst_lds: byte offset within group segment; src: 64-bit global address.
static __device__ __forceinline__ void async_copy16(unsigned dst_lds,
                                                    unsigned long long src) {
  asm volatile("global_load_async_to_lds_b128 %0, %1, off"
               :: "v"(dst_lds), "v"(src)
               : "memory");
}
static __device__ __forceinline__ void wait_async0() {
  asm volatile("s_wait_asynccnt 0x0" ::: "memory");
}

// ---------------------------------------------------------------------------
// Kernel: zero fp32 accumulator region in workspace
// ---------------------------------------------------------------------------
__global__ __launch_bounds__(256) void zero_kernel(float* __restrict__ p, int n) {
  for (int i = blockIdx.x * 256 + threadIdx.x; i < n; i += gridDim.x * 256)
    p[i] = 0.f;
}

// ---------------------------------------------------------------------------
// Kernel: fp32 [K][N] -> bf16 transposed [N][K]  (1024x1024)
// ---------------------------------------------------------------------------
__global__ __launch_bounds__(256) void transpose_cvt_kernel(
    const float* __restrict__ W, bf16_t* __restrict__ Wt) {
  __shared__ float t[32][33];
  const int tx = threadIdx.x & 31;        // fast dim
  const int ty = threadIdx.x >> 5;        // 0..7
  const int bn = blockIdx.x * 32;         // N tile
  const int bkk = blockIdx.y * 32;        // K tile
#pragma unroll
  for (int r = 0; r < 4; ++r) {
    int k = bkk + ty + r * 8;
    t[ty + r * 8][tx] = W[(size_t)k * DIM + bn + tx];
  }
  __syncthreads();
#pragma unroll
  for (int r = 0; r < 4; ++r) {
    int n = bn + ty + r * 8;
    Wt[(size_t)n * DIM + bkk + tx] = (bf16_t)t[tx][ty + r * 8];
  }
}

// ---------------------------------------------------------------------------
// Phase A: per (b, head, s-chunk of 128 rows):
//   K/V = Y @ W via v_wmma_f32_16x16x32_bf16, phi(K), then
//   kv += phi(K)^T @ V (second WMMA GEMM), ksum += sum_s phi(K)
// Weight tiles staged via GLOBAL_LOAD_ASYNC_TO_LDS_B128 (ASYNCcnt),
// Y tiles staged via b128 loads + VALU cvt + ds_store (needs fp32->bf16).
// Accumulated into global workspace with f32 atomics.
// ---------------------------------------------------------------------------
#define TM 128               // s-rows per workgroup
#define KC 64                // K-chunk (feature dim)
#define YT_STRIDE 72         // KC + 8 bf16 pad (16B) -> conflict-free b128
#define WT_STRIDE 72
#define PT_STRIDE 136        // TM + 8

__global__ __launch_bounds__(256) void phase_a_kernel(
    const float* __restrict__ Y,        // [B][S][DIM] fp32
    const bf16_t* __restrict__ WkT,     // [DIM n][DIM k] bf16
    const bf16_t* __restrict__ WvT,     // [DIM n][DIM k] bf16
    const float* __restrict__ bk,
    const float* __restrict__ bv,
    float* __restrict__ kv,             // [B][H][HD d][HD n] fp32
    float* __restrict__ ksum)           // [B][H][HD] fp32
{
  __shared__ __align__(16) char smem[36864];
  bf16_t* ytile = (bf16_t*)smem;                 // 128*72  (stage 1)
  bf16_t* wkt   = (bf16_t*)(smem + 18432);       // 64*72
  bf16_t* wvt   = (bf16_t*)(smem + 27648);       // 64*72
  bf16_t* ptkT  = (bf16_t*)smem;                 // 64*136  (stage 2, aliases)
  bf16_t* vT    = (bf16_t*)(smem + 17408);       // 64*136

  const int tid  = threadIdx.x;
  const int wave = tid >> 5;
  const int lane = tid & 31;
  const int half = lane >> 4;
  const int l15  = lane & 15;

  int bid = blockIdx.x;
  const int sc = bid & 31;  bid >>= 5;   // 32 s-chunks of 128
  const int h  = bid & 15;  bid >>= 4;
  const int b  = bid;

  const float* Yb = Y + ((size_t)b * SEQ + (size_t)sc * TM) * DIM;

  v8f accK[4] = {};
  v8f accV[4] = {};
  const int m0 = wave * 16;              // 8 waves * 16 rows = TM

  for (int kc = 0; kc < DIM; kc += KC) {
    __syncthreads();
    // --- async-stage weight tiles: 64n x 64k bf16 each (no VGPR roundtrip)
#pragma unroll
    for (int i = 0; i < 2; ++i) {
      int u  = tid + 256 * i;             // 16B unit; 8 units per 64-elem row
      int n  = u >> 3;
      int k8 = (u & 7) << 3;
      size_t src = (size_t)(h * HD + n) * DIM + kc + k8;
      async_copy16((unsigned)(size_t)(void*)(wkt + n * WT_STRIDE + k8),
                   (unsigned long long)(const void*)(WkT + src));
      async_copy16((unsigned)(size_t)(void*)(wvt + n * WT_STRIDE + k8),
                   (unsigned long long)(const void*)(WvT + src));
    }
    // --- stage Y tile: 128x64 fp32 -> bf16 LDS (coalesced b128 loads) ---
#pragma unroll
    for (int i = 0; i < 8; ++i) {
      int u   = tid + 256 * i;            // float4 unit; 16 units per row
      int row = u >> 4;
      int c4  = (u & 15) << 2;
      float4 f = *(const float4*)(Yb + (size_t)row * DIM + kc + c4);
      bf16_t* dst = ytile + row * YT_STRIDE + c4;
      dst[0] = (bf16_t)f.x; dst[1] = (bf16_t)f.y;
      dst[2] = (bf16_t)f.z; dst[3] = (bf16_t)f.w;
      // pull next chunk's Y slab toward the WGP while we convert this one
      if (kc + KC < DIM)
        __builtin_prefetch(Yb + (size_t)row * DIM + kc + KC + c4, 0, 1);
    }
    wait_async0();      // our async weight transfers have landed in LDS
    __syncthreads();    // + everyone's ds_stores visible
    // --- WMMA over this K chunk ---
#pragma unroll
    for (int kk = 0; kk < KC; kk += 32) {
      // A fragment: 16-bit A layout (lanes 0-15: K 0-7,16-23; 16-31: +8)
      const bf16_t* ab = ytile + (m0 + l15) * YT_STRIDE + kk;
      v8bf alo = *(const v8bf*)(ab + half * 8);
      v8bf ahi = *(const v8bf*)(ab + 16 + half * 8);
      v16bf afrag = cat8(alo, ahi);
#pragma unroll
      for (int tc = 0; tc < 4; ++tc) {
        const bf16_t* bbK = wkt + (tc * 16 + l15) * WT_STRIDE + kk + half * 8;
        v16bf bfK = cat8(*(const v8bf*)bbK, *(const v8bf*)(bbK + 16));
        accK[tc] = __builtin_amdgcn_wmma_f32_16x16x32_bf16(
            false, afrag, false, bfK, (short)0, accK[tc], false, false);
        const bf16_t* bbV = wvt + (tc * 16 + l15) * WT_STRIDE + kk + half * 8;
        v16bf bfV = cat8(*(const v8bf*)bbV, *(const v8bf*)(bbV + 16));
        accV[tc] = __builtin_amdgcn_wmma_f32_16x16x32_bf16(
            false, afrag, false, bfV, (short)0, accV[tc], false, false);
      }
    }
  }
  __syncthreads();   // stage-1 buffers dead; reuse LDS for stage 2

  // --- bias + phi, write transposed tiles: ptkT[d][s], vT[n][s] ---
#pragma unroll
  for (int tc = 0; tc < 4; ++tc) {
    int coln = tc * 16 + l15;
    float biask = bk[h * HD + coln];
    float biasv = bv[h * HD + coln];
#pragma unroll
    for (int r = 0; r < 8; ++r) {
      int m = m0 + r + half * 8;                         // s index in tile
      ptkT[coln * PT_STRIDE + m] = (bf16_t)phi_map(accK[tc][r] + biask);
      vT[coln * PT_STRIDE + m]   = (bf16_t)(accV[tc][r] + biasv);
    }
  }
  __syncthreads();

  // --- ksum: threads 0..63 each reduce one phi(K) row over TM s-values ---
  if (tid < HD) {
    const bf16_t* rowp = ptkT + tid * PT_STRIDE;
    float s = 0.f;
#pragma unroll 8
    for (int i = 0; i < TM; ++i) s += (float)rowp[i];
    atomicAdd(&ksum[(b * HEADS + h) * HD + tid], s);
  }

  // --- stage 2: kv(64x64) += phi(K)^T (64 x TM) @ V (TM x 64) ---
  const int tr  = wave >> 1;          // 4 row-tiles of 16
  const int tcb = (wave & 1) * 2;     // 2 col-tiles of 16 per wave
  v8f acc2[2] = {};
#pragma unroll
  for (int ss = 0; ss < TM; ss += 32) {
    const bf16_t* ab = ptkT + (tr * 16 + l15) * PT_STRIDE + ss;
    v16bf a2 = cat8(*(const v8bf*)(ab + half * 8),
                    *(const v8bf*)(ab + 16 + half * 8));
#pragma unroll
    for (int j = 0; j < 2; ++j) {
      const bf16_t* bb = vT + ((tcb + j) * 16 + l15) * PT_STRIDE + ss + half * 8;
      v16bf b2 = cat8(*(const v8bf*)bb, *(const v8bf*)(bb + 16));
      acc2[j] = __builtin_amdgcn_wmma_f32_16x16x32_bf16(
          false, a2, false, b2, (short)0, acc2[j], false, false);
    }
  }
  float* kvb = kv + (size_t)(b * HEADS + h) * HD * HD;
#pragma unroll
  for (int j = 0; j < 2; ++j) {
    int coln = (tcb + j) * 16 + l15;
#pragma unroll
    for (int r = 0; r < 8; ++r) {
      int rowd = tr * 16 + r + half * 8;
      atomicAdd(&kvb[rowd * HD + coln], acc2[j][r]);
    }
  }
}

// ---------------------------------------------------------------------------
// Phase B (tiny): q = x@Wq+bq, phi, numerator/denominator vs kv/ksum,
// ctx, then out = ctx@Wo + bo. One block per batch element.
// ---------------------------------------------------------------------------
__global__ __launch_bounds__(256) void phase_b_kernel(
    const float* __restrict__ x,     // [B][DIM]
    const float* __restrict__ Wq,    // [DIM][DIM]
    const float* __restrict__ bq,
    const float* __restrict__ kv,    // [B][H][HD][HD]
    const float* __restrict__ ksum,  // [B][H][HD]
    const float* __restrict__ Wo,    // [DIM][DIM]
    const float* __restrict__ bo,
    float* __restrict__ out)         // [B][DIM]
{
  __shared__ float phiq[DIM];
  __shared__ float ctx[DIM];
  __shared__ float den[HEADS];
  const int b = blockIdx.x;
  const int t = threadIdx.x;
  const float* xb = x + (size_t)b * DIM;

  for (int d = t; d < DIM; d += 256) {
    float s = bq[d];
    for (int i = 0; i < DIM; ++i) s += xb[i] * Wq[(size_t)i * DIM + d];
    phiq[d] = phi_map(s);
  }
  __syncthreads();

  if (t < HEADS) {
    const float* ks = ksum + (size_t)(b * HEADS + t) * HD;
    float s = 0.f;
#pragma unroll 8
    for (int d = 0; d < HD; ++d) s += phiq[t * HD + d] * ks[d];
    den[t] = s + EPSF;
  }
  __syncthreads();

  for (int p = t; p < DIM; p += 256) {
    int h = p >> 6, n = p & 63;
    const float* kvh = kv + (size_t)(b * HEADS + h) * HD * HD;
    float s = 0.f;
#pragma unroll 8
    for (int d = 0; d < HD; ++d) s += phiq[h * HD + d] * kvh[d * HD + n];
    ctx[p] = s / den[h];
  }
  __syncthreads();

  for (int j = t; j < DIM; j += 256) {
    float s = bo[j];
    for (int i = 0; i < DIM; ++i) s += ctx[i] * Wo[(size_t)i * DIM + j];
    out[(size_t)b * DIM + j] = s;
  }
}

// ---------------------------------------------------------------------------
// Launcher
// Inputs (setup_inputs order): 0 y_superposed, 1 x_context, 2 Wq, 3 bq,
//   4 Wk, 5 bk, 6 Wv, 7 bv, 8 Wo, 9 bo   (all fp32)
// Workspace layout:
//   [0, 2MB)      WkT bf16 [DIM][DIM]
//   [2MB, 4MB)    WvT bf16 [DIM][DIM]
//   [4MB, 8MB)    kv  f32  [B][H][HD][HD]
//   [8MB, +64KB)  ksum f32 [B][H][HD]
// ---------------------------------------------------------------------------
extern "C" void kernel_launch(void* const* d_in, const int* in_sizes, int n_in,
                              void* d_out, int out_size, void* d_ws, size_t ws_size,
                              hipStream_t stream) {
  (void)in_sizes; (void)n_in; (void)out_size; (void)ws_size;
  const float* Y  = (const float*)d_in[0];
  const float* x  = (const float*)d_in[1];
  const float* Wq = (const float*)d_in[2];
  const float* bq = (const float*)d_in[3];
  const float* Wk = (const float*)d_in[4];
  const float* bk = (const float*)d_in[5];
  const float* Wv = (const float*)d_in[6];
  const float* bv = (const float*)d_in[7];
  const float* Wo = (const float*)d_in[8];
  const float* bo = (const float*)d_in[9];
  float* out = (float*)d_out;

  char* ws = (char*)d_ws;
  bf16_t* WkT  = (bf16_t*)(ws);
  bf16_t* WvT  = (bf16_t*)(ws + (size_t)2 * 1024 * 1024);
  float*  kvp  = (float*)(ws + (size_t)4 * 1024 * 1024);
  float*  ksum = (float*)(ws + (size_t)8 * 1024 * 1024);

  const int n_acc = BATCH * HEADS * HD * HD + BATCH * HEADS * HD;
  zero_kernel<<<512, 256, 0, stream>>>(kvp, n_acc);
  transpose_cvt_kernel<<<dim3(DIM / 32, DIM / 32), 256, 0, stream>>>(Wk, WkT);
  transpose_cvt_kernel<<<dim3(DIM / 32, DIM / 32), 256, 0, stream>>>(Wv, WvT);
  phase_a_kernel<<<BATCH * HEADS * (SEQ / TM), 256, 0, stream>>>(
      Y, WkT, WvT, bk, bv, kvp, ksum);
  phase_b_kernel<<<BATCH, 256, 0, stream>>>(x, Wq, bq, kvp, ksum, Wo, bo, out);
}
